// ConditionalFeedForward_63324997812734
// MI455X (gfx1250) — compile-verified, matrix operand order checked by reading, not run on hardware
//
#include <hip/hip_runtime.h>
#include <hip/hip_bf16.h>

// ---------------- problem constants ----------------
#define Tt 4096
#define Dd 1024
#define Hh 2048
#define Ee 8
#define Aa 2
#define MT 32                 // token rows per workgroup tile (2 x 16 sub-tiles)
#define HC 128                // H chunk size (8 waves x 16)
#define MAX_SLOTS (Tt * Aa)   // 8192
#define TILES_PER_E (MAX_SLOTS / MT)  // 256

typedef __attribute__((ext_vector_type(16))) __bf16 v16bf;
typedef __attribute__((ext_vector_type(8)))  __bf16 v8bf;
typedef __attribute__((ext_vector_type(4)))  __bf16 v4bf;
typedef __attribute__((ext_vector_type(8)))  float  v8f;
typedef __attribute__((ext_vector_type(4)))  float  f32x4;

__device__ __forceinline__ v16bf make_frag(v8bf lo, v8bf hi) {
    return __builtin_shufflevector(lo, hi, 0, 1, 2, 3, 4, 5, 6, 7,
                                   8, 9, 10, 11, 12, 13, 14, 15);
}

// 8 contiguous fp32 -> 8 bf16 using native converts (v_cvt_pk_bf16_f32)
__device__ __forceinline__ v8bf cvt8(const float* __restrict__ p) {
    f32x4 a = *(const f32x4*)p;
    f32x4 b = *(const f32x4*)(p + 4);
    v8bf r;
#pragma unroll
    for (int i = 0; i < 4; ++i) {
        r[i]     = (__bf16)a[i];
        r[4 + i] = (__bf16)b[i];
    }
    return r;
}

// ISA 16-bit A/B fragment layout: lane holds K = {sel..sel+7, 16+sel..16+sel+7}
// relative to k0, sel = (lane<16 ? 0 : 8).  Row pointer is K-contiguous.
__device__ __forceinline__ v16bf lds_frag(const __bf16* __restrict__ row, int k0, int sel) {
    v8bf lo = *(const v8bf*)(row + k0 + sel);
    v8bf hi = *(const v8bf*)(row + k0 + 16 + sel);
    return make_frag(lo, hi);
}

__device__ __forceinline__ v16bf gbl_frag(const float* __restrict__ row, int k0, int sel) {
    return make_frag(cvt8(row + k0 + sel), cvt8(row + k0 + 16 + sel));
}

// fast SiLU(g)*u: v_exp_f32 + v_rcp_f32 (h is bf16-quantized anyway)
__device__ __forceinline__ float silu_mul(float g, float u) {
    float e = __builtin_amdgcn_exp2f(g * -1.44269504f);   // exp(-g)
    return g * __builtin_amdgcn_rcpf(1.0f + e) * u;
}

// ---------------- phase 0: bucket token-slots by expert ----------------
__global__ void moe_zero_counts(int* __restrict__ counts) {
    if (threadIdx.x < Ee) counts[threadIdx.x] = 0;
}

__global__ void moe_scatter(const int* __restrict__ expert_indices,
                            int* __restrict__ counts, int* __restrict__ lists) {
    int i = blockIdx.x * blockDim.x + threadIdx.x;
    if (i >= MAX_SLOTS) return;
    int e = expert_indices[i];
    int p = atomicAdd(&counts[e], 1);
    lists[e * MAX_SLOTS + p] = i;           // slot id = t*A + a
}

// ---------------- phase 1: fused grouped GEMM (gate/up -> silu -> down) ----------------
__global__ __launch_bounds__(256) void moe_ffn_kernel(
    const float* __restrict__ x,      // (T, D)
    const float* __restrict__ wg,     // (E, H, D)
    const float* __restrict__ wu,     // (E, H, D)
    const float* __restrict__ wd,     // (E, D, H)
    const int*   __restrict__ counts,
    const int*   __restrict__ lists,
    float*       __restrict__ out)    // (T, A, D) == slot-major
{
    __shared__ __bf16 xs[MT * Dd];    // 64 KB: 32 token rows of x in bf16
    __shared__ __bf16 hs[MT * HC];    // 8 KB:  h chunk (32 x 128) in bf16
    __shared__ int slots[MT];

    const int e    = blockIdx.x / TILES_PER_E;
    const int tile = blockIdx.x % TILES_PER_E;
    const int ne   = counts[e];
    if (tile * MT >= ne) return;      // uniform early exit

    const int tid  = threadIdx.x;
    const int lane = tid & 31;
    const int wave = tid >> 5;
    const int mcol = lane & 15;                 // A: M row / B: N col owned by lane
    const int sel  = (lane < 16) ? 0 : 8;       // K sub-block selector
    const int mhi  = (lane < 16) ? 0 : 8;       // C/D: +8 row offset for hi lanes

    if (tid < MT) {
        int r = tile * MT + tid;
        slots[tid] = lists[e * MAX_SLOTS + (r < ne ? r : tile * MT)]; // pad = dup row
    }
    __syncthreads();

    // stage x tile -> LDS bf16 (16B global loads, 8B packed LDS stores)
    for (int i = tid; i < (MT * Dd) / 4; i += 256) {
        int idx = i * 4;
        int m = idx >> 10;                       // / Dd
        int k = idx & (Dd - 1);
        f32x4 v = *(const f32x4*)(x + (size_t)(slots[m] / Aa) * Dd + k);
        v4bf r;
#pragma unroll
        for (int j = 0; j < 4; ++j) r[j] = (__bf16)v[j];
        *(v4bf*)(xs + idx) = r;
    }
    __syncthreads();

    const size_t wgate_base = (size_t)e * Hh * Dd;
    const size_t wdown_base = (size_t)e * Dd * Hh;

    // persistent output accumulators: 2 M sub-tiles x 8 D tiles (128 VGPRs)
    v8f acc[2][8];
#pragma unroll
    for (int s = 0; s < 2; ++s)
#pragma unroll
        for (int j = 0; j < 8; ++j)
#pragma unroll
            for (int r = 0; r < 8; ++r) acc[s][j][r] = 0.0f;

    const __bf16* __restrict__ xrow0 = xs + mcol * Dd;
    const __bf16* __restrict__ xrow1 = xs + (16 + mcol) * Dd;

    for (int chunk = 0; chunk < Hh / HC; ++chunk) {
        // ---- GEMM1: this wave computes gate & up for h cols [chunk*HC + wave*16, +16)
        const int hrow = chunk * HC + wave * 16 + mcol;     // B column owned by lane
        const float* __restrict__ wgp = wg + wgate_base + (size_t)hrow * Dd;
        const float* __restrict__ wup = wu + wgate_base + (size_t)hrow * Dd;

        v8f ag0, au0, ag1, au1;
#pragma unroll
        for (int r = 0; r < 8; ++r) { ag0[r] = au0[r] = ag1[r] = au1[r] = 0.0f; }

#pragma unroll 2
        for (int k0 = 0; k0 < Dd; k0 += 32) {
            v16bf a0 = lds_frag(xrow0, k0, sel);
            v16bf a1 = lds_frag(xrow1, k0, sel);
            v16bf bg = gbl_frag(wgp, k0, sel);
            v16bf bu = gbl_frag(wup, k0, sel);
            ag0 = __builtin_amdgcn_wmma_f32_16x16x32_bf16(false, a0, false, bg,
                                                          (short)0, ag0, false, false);
            au0 = __builtin_amdgcn_wmma_f32_16x16x32_bf16(false, a0, false, bu,
                                                          (short)0, au0, false, false);
            ag1 = __builtin_amdgcn_wmma_f32_16x16x32_bf16(false, a1, false, bg,
                                                          (short)0, ag1, false, false);
            au1 = __builtin_amdgcn_wmma_f32_16x16x32_bf16(false, a1, false, bu,
                                                          (short)0, au1, false, false);
        }

        // silu(g) * u -> bf16 h chunk (C/D layout: row = r + mhi, col = mcol)
#pragma unroll
        for (int r = 0; r < 8; ++r) {
            hs[(r + mhi) * HC + wave * 16 + mcol]      = (__bf16)silu_mul(ag0[r], au0[r]);
            hs[(16 + r + mhi) * HC + wave * 16 + mcol] = (__bf16)silu_mul(ag1[r], au1[r]);
        }
        __syncthreads();

        // ---- GEMM2: accumulate out tiles over K = this h chunk (128)
        const __bf16* __restrict__ hrow0 = hs + mcol * HC;
        const __bf16* __restrict__ hrow1 = hs + (16 + mcol) * HC;
#pragma unroll
        for (int k0 = 0; k0 < HC; k0 += 32) {
            v16bf a0 = lds_frag(hrow0, k0, sel);
            v16bf a1 = lds_frag(hrow1, k0, sel);
#pragma unroll
            for (int j = 0; j < 8; ++j) {
                int d = (wave * 8 + j) * 16 + mcol;         // B column = output dim
                const float* __restrict__ wdp =
                    wd + wdown_base + (size_t)d * Hh + chunk * HC;
                v16bf bf = gbl_frag(wdp, k0, sel);
                acc[0][j] = __builtin_amdgcn_wmma_f32_16x16x32_bf16(
                    false, a0, false, bf, (short)0, acc[0][j], false, false);
                acc[1][j] = __builtin_amdgcn_wmma_f32_16x16x32_bf16(
                    false, a1, false, bf, (short)0, acc[1][j], false, false);
            }
        }
        __syncthreads();   // protect hs before next chunk overwrites it
    }

    // ---- store: out[slot, :] for valid rows
#pragma unroll
    for (int s = 0; s < 2; ++s) {
#pragma unroll
        for (int j = 0; j < 8; ++j) {
            int d = (wave * 8 + j) * 16 + mcol;
#pragma unroll
            for (int r = 0; r < 8; ++r) {
                int m = s * 16 + r + mhi;
                if (tile * MT + m < ne)
                    out[(size_t)slots[m] * Dd + d] = acc[s][j][r];
            }
        }
    }
}

extern "C" void kernel_launch(void* const* d_in, const int* in_sizes, int n_in,
                              void* d_out, int out_size, void* d_ws, size_t ws_size,
                              hipStream_t stream) {
    const float* x  = (const float*)d_in[0];
    const int*   ei = (const int*)d_in[1];      // expert_indices (integer input)
    const float* wg = (const float*)d_in[2];
    const float* wu = (const float*)d_in[3];
    const float* wd = (const float*)d_in[4];
    float* out = (float*)d_out;

    int* counts = (int*)d_ws;                   // E ints
    int* lists  = counts + 64;                  // 256 B offset; E*8192 ints (~256 KB)

    moe_zero_counts<<<1, 32, 0, stream>>>(counts);
    moe_scatter<<<(MAX_SLOTS + 255) / 256, 256, 0, stream>>>(ei, counts, lists);
    moe_ffn_kernel<<<Ee * TILES_PER_E, 256, 0, stream>>>(x, wg, wu, wd, counts, lists, out);
}
